// MultiHeadAttention_65515431133521
// MI455X (gfx1250) — matrix operand order repaired and checked by reading, not compiled, not run on hardware
//
#include <hip/hip_runtime.h>

// MI455X / gfx1250 causal MHA forward.  B=2, L=2048, D=1024, H=16, HD=64.
// ~60 GFLOP vs ~50MB of tensors (all L2-resident): compute-bound on the WMMA
// pipes. Pipeline: (1) one-shot f32->bf16 conversion of x/Wqkv/Wo,
// (2) QKV GEMM with TDM (tensor_load_to_lds) double-buffered LDS staging,
// (3) wave-per-tile flash attention, (4) out-proj GEMM -> f32.

typedef unsigned short ushort_t;
typedef __attribute__((ext_vector_type(16))) __bf16 v16bf;
typedef __attribute__((ext_vector_type(8)))  float  v8f;
typedef __attribute__((ext_vector_type(4)))  unsigned int u32x4;
typedef __attribute__((ext_vector_type(4)))  int          i32x4;
typedef __attribute__((ext_vector_type(8)))  int          i32x8;

#define LDW 40            // LDS row stride in halfs: 80B (64B row + 16B pad)
#define NEG_BIG (-3.0e38f)

#if defined(__has_builtin)
#if __has_builtin(__builtin_amdgcn_tensor_load_to_lds)
#define HAVE_TDM 1
#endif
#endif
#ifndef HAVE_TDM
#define HAVE_TDM 0
#endif

__device__ __forceinline__ ushort_t f2bf(float f) {
    union { float f; unsigned u; } v; v.f = f;
    unsigned r = v.u + 0x7FFFu + ((v.u >> 16) & 1u);  // round-to-nearest-even
    return (ushort_t)(r >> 16);
}

struct U8x32 { uint4 lo, hi; };
__device__ __forceinline__ v16bf mkfrag(uint4 lo, uint4 hi) {
    U8x32 t{lo, hi};
    return __builtin_bit_cast(v16bf, t);   // VGPR0-3 = lo, VGPR4-7 = hi
}

__device__ __forceinline__ v8f wmma_bf16(v16bf a, v16bf b, v8f c) {
    return __builtin_amdgcn_wmma_f32_16x16x32_bf16(false, a, false, b, (short)0, c,
                                                   false, false);
}

__device__ __forceinline__ float half16_max(float v) {
    v = fmaxf(v, __shfl_xor(v, 1, 32));
    v = fmaxf(v, __shfl_xor(v, 2, 32));
    v = fmaxf(v, __shfl_xor(v, 4, 32));
    v = fmaxf(v, __shfl_xor(v, 8, 32));
    return v;
}
__device__ __forceinline__ float half16_sum(float v) {
    v += __shfl_xor(v, 1, 32);
    v += __shfl_xor(v, 2, 32);
    v += __shfl_xor(v, 4, 32);
    v += __shfl_xor(v, 8, 32);
    return v;
}

// ---------------------------------------------------------------------------
// TDM: async 2-D bf16 tile (tileD0 x tileD1) from global -> LDS, with LDS
// padding of 16B after every 64B row (matches LDW=40 halfs).
// ---------------------------------------------------------------------------
__device__ __forceinline__ void tdm_load_2d(unsigned long long gaddr, unsigned ldsOff,
                                            unsigned tensorD0, unsigned tensorD1,
                                            unsigned stride0,
                                            unsigned tileD0, unsigned tileD1) {
#if HAVE_TDM
    u32x4 g0;
    g0[0] = 1u;                                            // count=1, user D#
    g0[1] = ldsOff;                                        // lds_addr (bytes)
    g0[2] = (unsigned)(gaddr & 0xFFFFFFFFull);             // global_addr[31:0]
    g0[3] = (unsigned)((gaddr >> 32) & 0x01FFFFFFull)      // global_addr[56:32]
            | 0x80000000u;                                 // type=2 (image)
    i32x8 g1;
    g1[0] = (int)((1u << 16)      // data_size = 2 bytes
                | (1u << 20)      // pad_enable
                | (3u << 22)      // pad_interval: 16 DWORDs (64B)
                | (3u << 25));    // pad_amount:   4 DWORDs (16B)
    g1[1] = (int)((tensorD0 & 0xFFFFu) << 16);
    g1[2] = (int)(((tensorD0 >> 16) & 0xFFFFu) | ((tensorD1 & 0xFFFFu) << 16));
    g1[3] = (int)(((tensorD1 >> 16) & 0xFFFFu) | ((tileD0 & 0xFFFFu) << 16));
    g1[4] = (int)(tileD1 & 0xFFFFu);                       // tile_dim1 (tile_dim2=0)
    g1[5] = (int)stride0;                                  // tensor_dim0_stride lo
    g1[6] = 0; g1[7] = 0;
    i32x4 z4 = {0, 0, 0, 0};
#if __clang_major__ >= 23
    i32x8 z8 = {0, 0, 0, 0, 0, 0, 0, 0};
    __builtin_amdgcn_tensor_load_to_lds(g0, g1, z4, z4, z8, 0);
#else
    __builtin_amdgcn_tensor_load_to_lds(g0, g1, z4, z4, 0);
#endif
#else
    (void)gaddr; (void)ldsOff; (void)tensorD0; (void)tensorD1;
    (void)stride0; (void)tileD0; (void)tileD1;
#endif
}

// ---------------------------------------------------------------------------
// One-shot f32 -> bf16 conversion (bandwidth-trivial at 23.3 TB/s).
// ---------------------------------------------------------------------------
__global__ __launch_bounds__(256)
void cvt_bf16(const float* __restrict__ in, ushort_t* __restrict__ out, int n4) {
    int i = blockIdx.x * 256 + threadIdx.x;
    if (i < n4) {
        float4 q = ((const float4*)in)[i];
        uint2 r;
        r.x = (unsigned)f2bf(q.x) | ((unsigned)f2bf(q.y) << 16);
        r.y = (unsigned)f2bf(q.z) | ((unsigned)f2bf(q.w) << 16);
        ((uint2*)out)[i] = r;
    }
}

// ---------------------------------------------------------------------------
// Tiled bf16 GEMM: C[M,N] = A[M,K] @ B[N,K]^T.  128x128x32 block, 8 waves
// (4Mx2N), wave sub-tile 32x64 = 2x4 WMMA accs.  A/B tiles staged by the TDM
// into double-buffered LDS; wave 0 drives the DMA, s_wait_tensorcnt + barrier
// publishes; compute on buffer p overlaps the DMA filling p^1.
// ---------------------------------------------------------------------------
template<typename TOut, bool WRITE_VT>
__global__ __launch_bounds__(256)
void gemm_nt(const ushort_t* __restrict__ A, const ushort_t* __restrict__ B,
             TOut* __restrict__ C, ushort_t* __restrict__ vT,
             int M, int N, int K) {
    __shared__ __align__(16) ushort_t smem[2 * 2 * 128 * LDW];  // 40KB

    const int t    = threadIdx.x;
    const int lane = t & 31;
    const int wv   = t >> 5;
    const int wM   = wv & 3;
    const int wN   = wv >> 2;
    const int lm   = lane & 15;
    const int lh   = lane >> 4;
    const int m0   = blockIdx.y * 128;
    const int n0   = blockIdx.x * 128;

    v8f acc[2][4] = {};

#if HAVE_TDM
    const unsigned long long aBase = (unsigned long long)(const void*)A;
    const unsigned long long bBase = (unsigned long long)(const void*)B;
    auto issue_pair = [&](int k0, int p) {
        unsigned aOff = (unsigned)(unsigned long long)(const void*)(smem + p * (2 * 128 * LDW));
        unsigned bOff = aOff + 128 * LDW * 2u;
        tdm_load_2d(aBase + ((unsigned long long)m0 * K + k0) * 2ull, aOff,
                    (unsigned)K, (unsigned)M, (unsigned)K, 32u, 128u);
        tdm_load_2d(bBase + ((unsigned long long)n0 * K + k0) * 2ull, bOff,
                    (unsigned)K, (unsigned)N, (unsigned)K, 32u, 128u);
    };
    if (wv == 0) issue_pair(0, 0);
    int pb = 0;
    for (int k0 = 0; k0 < K; k0 += 32) {
        if (wv == 0) {
            if (k0 + 32 < K) {
                issue_pair(k0 + 32, pb ^ 1);
                __builtin_amdgcn_s_wait_tensorcnt(2);   // current pair done
            } else {
                __builtin_amdgcn_s_wait_tensorcnt(0);   // last pair done
            }
        }
        __syncthreads();
        const ushort_t* As = smem + pb * (2 * 128 * LDW);
        const ushort_t* Bs = As + 128 * LDW;
#else
    for (int k0 = 0; k0 < K; k0 += 32) {
        ushort_t* AsW = smem;
        ushort_t* BsW = smem + 128 * LDW;
        for (int i = 0; i < 4; ++i) {            // manual bf16 tile copy fallback
            int f = t + 256 * i, r = f >> 3, c4 = f & 7;
            *(uint2*)&AsW[r * LDW + c4 * 4] =
                *(const uint2*)(A + (size_t)(m0 + r) * K + k0 + c4 * 4);
            *(uint2*)&BsW[r * LDW + c4 * 4] =
                *(const uint2*)(B + (size_t)(n0 + r) * K + k0 + c4 * 4);
        }
        __syncthreads();
        const ushort_t* As = smem;
        const ushort_t* Bs = smem + 128 * LDW;
#endif
        // A 16x32 frag: VGPR0-3 K = lh*8+0..7 ; VGPR4-7 K = 16+lh*8+0..7
        v16bf af[2];
#pragma unroll
        for (int mi = 0; mi < 2; ++mi) {
            const ushort_t* p = &As[(wM * 32 + mi * 16 + lm) * LDW];
            af[mi] = mkfrag(*(const uint4*)(p + lh * 8),
                            *(const uint4*)(p + 16 + lh * 8));
        }
        // B 32x16 frag: N = lm ; VGPR0-3 K = lh*8+0..7 ; VGPR4-7 K = 16+lh*8+0..7
        v16bf bfv[4];
#pragma unroll
        for (int ni = 0; ni < 4; ++ni) {
            const ushort_t* p = &Bs[(wN * 64 + ni * 16 + lm) * LDW];
            bfv[ni] = mkfrag(*(const uint4*)(p + lh * 8),
                             *(const uint4*)(p + 16 + lh * 8));
        }
#pragma unroll
        for (int mi = 0; mi < 2; ++mi)
#pragma unroll
            for (int ni = 0; ni < 4; ++ni)
                acc[mi][ni] = wmma_bf16(af[mi], bfv[ni], acc[mi][ni]);
        __syncthreads();
#if HAVE_TDM
        pb ^= 1;
#endif
    }

    // Epilogue: C layout row = v + 8*lh, col = lm.
#pragma unroll
    for (int mi = 0; mi < 2; ++mi)
#pragma unroll
        for (int ni = 0; ni < 4; ++ni)
#pragma unroll
            for (int v = 0; v < 8; ++v) {
                int gR = m0 + wM * 32 + mi * 16 + v + 8 * lh;
                int gC = n0 + wN * 64 + ni * 16 + lm;
                float val = acc[mi][ni][v];
                if constexpr (sizeof(TOut) == 4) {
                    C[(size_t)gR * N + gC] = val;
                } else {
                    C[(size_t)gR * N + gC] = (TOut)f2bf(val);
                }
                if constexpr (WRITE_VT) {
                    if (gC >= 2048) {               // V third of qkv (2D = 2048)
                        int e  = gC - 2048;
                        int hh = e >> 6, dd = e & 63;
                        int bb = gR >> 11, ll = gR & 2047;
                        vT[((size_t)((bb * 16 + hh) * 64 + dd)) * 2048 + ll] = f2bf(val);
                    }
                }
            }
}

// ---------------------------------------------------------------------------
// Flash attention: one wave per 16-query tile, 4 independent waves / block.
// ---------------------------------------------------------------------------
__global__ __launch_bounds__(128)
void attn_kernel(const ushort_t* __restrict__ qkv,   // [B*L, 3072] bf16
                 const ushort_t* __restrict__ vT,    // [B,H,64,L]  bf16
                 ushort_t* __restrict__ attnO) {     // [B*L, 1024] bf16
    __shared__ __align__(16) ushort_t pLds[4 * 16 * LDW];

    const int wave = threadIdx.x >> 5;
    const int lane = threadIdx.x & 31;
    const int lm   = lane & 15;
    const int lh   = lane >> 4;
    const int b    = blockIdx.z;
    const int h    = blockIdx.y;
    const int qt   = blockIdx.x * 4 + wave;
    const int qBase = qt * 16;

    const size_t rowQ = (size_t)(b * 2048 + qBase + lm) * 3072 + h * 64;
    v16bf aQ[2];
#pragma unroll
    for (int c = 0; c < 2; ++c)
        aQ[c] = mkfrag(*(const uint4*)(qkv + rowQ + c * 32 + lh * 8),
                       *(const uint4*)(qkv + rowQ + c * 32 + 16 + lh * 8));

    float mrow[8], lrow[8];
#pragma unroll
    for (int v = 0; v < 8; ++v) { mrow[v] = NEG_BIG; lrow[v] = 0.0f; }
    v8f o[4] = {};

    const size_t vtBase = ((size_t)((b * 16 + h) * 64)) * 2048;
    ushort_t* myP = &pLds[wave * 16 * LDW];

    const int nB = (qBase + 15) / 32 + 1;
    for (int kb = 0; kb < nB; ++kb) {
        const int kBase = kb * 32;

        v8f s[2];
#pragma unroll
        for (int j = 0; j < 2; ++j) {
            const size_t rowK = (size_t)(b * 2048 + kBase + j * 16 + lm) * 3072 + 1024 + h * 64;
            v16bf bk0 = mkfrag(*(const uint4*)(qkv + rowK + lh * 8),
                               *(const uint4*)(qkv + rowK + 16 + lh * 8));
            v16bf bk1 = mkfrag(*(const uint4*)(qkv + rowK + 32 + lh * 8),
                               *(const uint4*)(qkv + rowK + 32 + 16 + lh * 8));
            v8f c = {};
            c = wmma_bf16(aQ[0], bk0, c);
            c = wmma_bf16(aQ[1], bk1, c);
            s[j] = c;
        }

        const bool needMask = (kBase + 31 > qBase);

#pragma unroll
        for (int v = 0; v < 8; ++v) {
            float s0 = s[0][v] * 0.125f;       // 1/sqrt(64)
            float s1 = s[1][v] * 0.125f;
            if (needMask) {
                int qg = qBase + v + 8 * lh;
                if (kBase + lm > qg)      s0 = NEG_BIG;
                if (kBase + 16 + lm > qg) s1 = NEG_BIG;
            }
            float mx = half16_max(fmaxf(s0, s1));
            float mn = fmaxf(mrow[v], mx);
            s0 = __expf(s0 - mn);
            s1 = __expf(s1 - mn);
            float sum = half16_sum(s0 + s1);
            float al  = __expf(mrow[v] - mn);
            lrow[v] = lrow[v] * al + sum;
            mrow[v] = mn;
            o[0][v] *= al; o[1][v] *= al; o[2][v] *= al; o[3][v] *= al;
            ushort_t* pr = &myP[(v + 8 * lh) * LDW];
            pr[lm]      = f2bf(s0);
            pr[16 + lm] = f2bf(s1);
        }

        const ushort_t* pp = &myP[lm * LDW];
        v16bf aP = mkfrag(*(const uint4*)(pp + lh * 8),
                          *(const uint4*)(pp + 16 + lh * 8));

#pragma unroll
        for (int j = 0; j < 4; ++j) {
            const ushort_t* vr = vT + vtBase + (size_t)(j * 16 + lm) * 2048 + kBase;
            v16bf bv = mkfrag(*(const uint4*)(vr + lh * 8),
                              *(const uint4*)(vr + 16 + lh * 8));
            o[j] = wmma_bf16(aP, bv, o[j]);
        }
    }

#pragma unroll
    for (int v = 0; v < 8; ++v) {
        float inv = 1.0f / lrow[v];
        size_t row = (size_t)(b * 2048 + qBase + v + 8 * lh) * 1024 + h * 64;
#pragma unroll
        for (int j = 0; j < 4; ++j)
            attnO[row + j * 16 + lm] = f2bf(o[j][v] * inv);
    }
}

// ---------------------------------------------------------------------------
extern "C" void kernel_launch(void* const* d_in, const int* in_sizes, int n_in,
                              void* d_out, int out_size, void* d_ws, size_t ws_size,
                              hipStream_t stream) {
    const float* x    = (const float*)d_in[0];   // [2,2048,1024]
    const float* Wqkv = (const float*)d_in[1];   // [3072,1024]
    const float* Wo   = (const float*)d_in[2];   // [1024,1024]
    float* out = (float*)d_out;                  // f32

    // ws (bf16): xb 8MB | Wqkv 6MB | Wo 2MB | qkv 24MB | vT 8MB | attn 8MB
    ushort_t* xb   = (ushort_t*)d_ws;
    ushort_t* wqb  = xb  + (size_t)4096 * 1024;
    ushort_t* wob  = wqb + (size_t)3072 * 1024;
    ushort_t* qkv  = wob + (size_t)1024 * 1024;
    ushort_t* vT   = qkv + (size_t)4096 * 3072;
    ushort_t* attn = vT  + (size_t)2 * 16 * 64 * 2048;

    // 0) one-shot bf16 conversion of all f32 operands
    cvt_bf16<<<4096, 256, 0, stream>>>(x,    xb,  4096 * 1024 / 4);
    cvt_bf16<<<3072, 256, 0, stream>>>(Wqkv, wqb, 3072 * 1024 / 4);
    cvt_bf16<<<1024, 256, 0, stream>>>(Wo,   wob, 1024 * 1024 / 4);

    // 1) qkv = x @ Wqkv^T  (TDM-staged bf16 GEMM; V also scattered transposed)
    gemm_nt<ushort_t, true>
        <<<dim3(3072 / 128, 4096 / 128), 256, 0, stream>>>(xb, wqb, qkv, vT,
                                                           4096, 3072, 1024);
    // 2) causal flash attention
    attn_kernel<<<dim3(2048 / 16 / 4, 16, 2), 128, 0, stream>>>(qkv, vT, attn);
    // 3) out = attn @ Wo^T  (bf16 in, f32 out)
    gemm_nt<float, false>
        <<<dim3(1024 / 128, 4096 / 128), 256, 0, stream>>>(attn, wob, out, nullptr,
                                                           4096, 1024, 1024);
}
